// OC_RSSM_42348377538712
// MI455X (gfx1250) — matrix operand order repaired and checked by reading, not compiled, not run on hardware
//
#include <hip/hip_runtime.h>
#include <math.h>

// ============================================================================
// OC_RSSM greedy autoregressive decode for MI455X (gfx1250).
//
// Latency-bound serial 256-step chain => single persistent workgroup
// (8 wave32s on one WGP):
//  - KV-cached incremental decode (identical math to the reference's full
//    causal recompute).
//  - All GEMMs on v_wmma_f32_16x16x32_bf16 (M=16 == batch, f32 accumulate).
//  - Weights converted once to bf16, stored transposed [N][K] so each lane's
//    B fragment is one contiguous 32B load.
//  - A fragments (shared activations) are hoisted out of the N-tile loops:
//    built once per wave per GEMM phase instead of once per tile.
//  - Skinny attention (1 token x <=257 keys x 16 dims) = online-softmax VALU
//    loop over a bf16 KV cache (L2-resident).
// ============================================================================

typedef __bf16 bf16_t;
typedef __attribute__((ext_vector_type(16))) __bf16 v16bf;
typedef __attribute__((ext_vector_type(8)))  float  v8f;

#define NB  2     // transformer blocks
#define BB  16    // batch
#define DM  64    // d_model
#define NH  4     // heads
#define DH  16    // head dim
#define FF  256   // ffn dim
#define NV  2048  // vocab
#define GL  256   // generated tokens
#define LS  257   // max sequence (BOS + GL)
#define NWAVES 8

// bf16 weight arena element offsets; per-block stride 65536 elements
#define OFF_WQ   0
#define OFF_WK   4096
#define OFF_WV   8192
#define OFF_WO   12288
#define OFF_CWQ  16384
#define OFF_CWK  20480
#define OFF_CWV  24576
#define OFF_CWO  28672
#define OFF_FC1  32768   // [256][64]
#define OFF_FC2  49152   // [64][256]
#define BLK_STRIDE 65536
#define OFF_WOUT 131072  // [2048][64]

// ---------------------------------------------------------------------------
// WMMA fragment helpers (wave32, 16x16x32 bf16)
// A (16x32, 16-bit): lane = row (mod 16); lanes 0-15 hold K 0-15,
// lanes 16-31 hold K 16-31 (same k order as B).
// ---------------------------------------------------------------------------
__device__ inline v16bf frag_a(const float* __restrict__ a, int ld, int kb, int lane) {
  const int row = lane & 15;
  const int kh  = (lane >> 4) << 4;     // 0 or 16
  v16bf f;
#pragma unroll
  for (int e = 0; e < 16; ++e) {
    const int k = kh + e;
    f[e] = (bf16_t)a[row * ld + kb + k];
  }
  return f;
}

// Build all K/32 A fragments for a 16xK activation once (loop-invariant
// across N-tiles; executed uniformly by every thread in the wave).
template <int K>
__device__ inline void load_afrags(const float* __restrict__ a, int lane, v16bf* af) {
#pragma unroll
  for (int c = 0; c < K / 32; ++c) af[c] = frag_a(a, K, c * 32, lane);
}

// B (32x16): lane = column (mod 16); lanes 0-15 hold K 0-15, lanes 16-31 hold
// K 16-31, contiguous per lane => one aligned 32B load from W^T[N][K].
__device__ inline v16bf frag_b(const bf16_t* __restrict__ wT, int K, int nt, int kb, int lane) {
  const bf16_t* p = wT + (size_t)(nt * 16 + (lane & 15)) * K + kb + ((lane >> 4) << 4);
  return *(const v16bf*)p;   // 32B-aligned by construction
}

// One 16x16 output tile from preloaded A fragments.
template <int K>
__device__ inline v8f gemm_tile_pre(const v16bf* __restrict__ af,
                                    const bf16_t* __restrict__ wT,
                                    int nt, int lane) {
  v8f d = {};
#pragma unroll
  for (int c = 0; c < K / 32; ++c) {
    v16bf bf = frag_b(wT, K, nt, c * 32, lane);
    d = __builtin_amdgcn_wmma_f32_16x16x32_bf16(false, af[c], false, bf,
                                                (short)0, d, false, false);
  }
  return d;
}

// C/D layout: VGPR r -> row (r + 8*(lane>=16)), col = lane&15 (ISA 7.12.2)
__device__ inline void store_tile(float* y, int ldn, int nt, int lane, v8f d, float mult) {
  const int col = nt * 16 + (lane & 15);
  const int rb  = (lane >> 4) << 3;
#pragma unroll
  for (int r = 0; r < 8; ++r) y[(rb + r) * ldn + col] = d[r] * mult;
}
__device__ inline void store_tile_add(float* y, int ldn, int nt, int lane, v8f d) {
  const int col = nt * 16 + (lane & 15);
  const int rb  = (lane >> 4) << 3;
#pragma unroll
  for (int r = 0; r < 8; ++r) y[(rb + r) * ldn + col] += d[r];
}
__device__ inline void store_tile_bias_relu(float* y, int ldn, int nt, int lane, v8f d,
                                            const float* __restrict__ bias) {
  const int col = nt * 16 + (lane & 15);
  const int rb  = (lane >> 4) << 3;
  const float bv = bias[col];
#pragma unroll
  for (int r = 0; r < 8; ++r) y[(rb + r) * ldn + col] = fmaxf(d[r] + bv, 0.0f);
}
__device__ inline void store_tile_add_bias(float* y, int ldn, int nt, int lane, v8f d,
                                           const float* __restrict__ bias) {
  const int col = nt * 16 + (lane & 15);
  const int rb  = (lane >> 4) << 3;
  const float bv = bias[col];
#pragma unroll
  for (int r = 0; r < 8; ++r) y[(rb + r) * ldn + col] += d[r] + bv;
}

__device__ inline void row_ln(const float* __restrict__ xin, float* __restrict__ xout,
                              const float* __restrict__ g, const float* __restrict__ b,
                              int tid) {
  if (tid < BB) {
    const float* r = xin + tid * DM;
    float m = 0.f;
#pragma unroll
    for (int d = 0; d < DM; ++d) m += r[d];
    m *= (1.0f / DM);
    float v = 0.f;
#pragma unroll
    for (int d = 0; d < DM; ++d) { float t = r[d] - m; v += t * t; }
    v *= (1.0f / DM);
    const float inv = rsqrtf(v + 1e-5f);
    float* o = xout + tid * DM;
#pragma unroll
    for (int d = 0; d < DM; ++d) o[d] = (r[d] - m) * inv * g[d] + b[d];
  }
}

// ---------------------------------------------------------------------------
// Small utility kernels
// ---------------------------------------------------------------------------
__global__ void oc_zero_f32(float* __restrict__ p, int n) {
  const int i = blockIdx.x * blockDim.x + threadIdx.x;
  if (i < n) p[i] = 0.0f;
}

// dst[n*K + k] = (bf16) src[k*N + n]   (build W^T in bf16)
__global__ void oc_wt_transpose_bf16(const float* __restrict__ src,
                                     bf16_t* __restrict__ dst, int K, int N) {
  const int i = blockIdx.x * blockDim.x + threadIdx.x;
  if (i >= K * N) return;
  const int n = i / K, k = i - n * K;
  dst[i] = (bf16_t)src[k * N + n];
}

// ---------------------------------------------------------------------------
// Persistent decode kernel (single workgroup, 8 waves)
// ---------------------------------------------------------------------------
struct KP {
  const float* slots;        // (16,8,64)
  const float* dict;         // (2049,64)
  const float* pe;           // (257,64)
  const float* lng[7];       // ln1/ln2/ln3 per block, [6]=final
  const float* lnb[7];
  const float* fc1b[2];
  const float* fc2b[2];
  const bf16_t* wT;          // bf16 transposed-weight arena
  bf16_t* kcache;            // [blk][b*4+h][pos][16] bf16
  bf16_t* vcache;
  float*  kcross;            // [blk][b*4+h][slot][16] f32
  float*  vcross;
  float*  logits;            // [16][2048] f32 scratch
  float*  out_z;             // (B,VOCAB,16,16)
  float*  out_logp;          // (B,GL,VOCAB)
};

__global__ __launch_bounds__(256, 1) void oc_rssm_decode(KP P) {
  const int tid  = threadIdx.x;
  const int lane = tid & 31;
  const int wave = tid >> 5;

  __shared__ float s_x[BB * DM];        // residual stream
  __shared__ float s_h[BB * DM];        // LN output / attention output
  __shared__ float s_q[BB * DM];
  __shared__ float s_k[BB * DM];
  __shared__ float s_v[BB * DM];
  __shared__ float s_big[128 * DM];     // init: slots (128x64); steps: FFN mid (16x256)
  __shared__ float s_red[256];
  __shared__ int   s_redi[256];
  __shared__ float s_rowm[BB], s_rowl[BB];
  __shared__ int   s_rowi[BB], s_tok[BB];

  // ---- init: stage slots, precompute cross-attention K/V (WMMA) ----
  for (int u = tid; u < 128 * DM; u += 256) s_big[u] = P.slots[u];
  if (tid < BB) s_tok[tid] = 0;          // BOS id 0
  __syncthreads();

  // 2 blk x {K,V} x 8 M-tiles x 4 N-tiles = 128 tiles, wave-strided (uniform)
  for (int tt = wave; tt < 128; tt += NWAVES) {
    const int blk = tt >> 6, kv = (tt >> 5) & 1, mt = (tt >> 2) & 7, nt = tt & 3;
    const bf16_t* w = P.wT + (size_t)blk * BLK_STRIDE + (kv ? OFF_CWV : OFF_CWK);
    v16bf af[2];
    load_afrags<DM>(s_big + mt * 16 * DM, lane, af);   // A differs per M-tile here
    v8f d = gemm_tile_pre<DM>(af, w, nt, lane);
    const int col = nt * 16 + (lane & 15), rb = (lane >> 4) << 3;
    float* dst = kv ? P.vcross : P.kcross;
#pragma unroll
    for (int r = 0; r < 8; ++r) {
      const int row = mt * 16 + rb + r;      // row = b*8 + slot
      const int b8 = row >> 3, sl = row & 7, hh = col >> 4, dd = col & 15;
      dst[(((size_t)blk * 64 + b8 * NH + hh) * 8 + sl) * DH + dd] = d[r];
    }
  }
  __threadfence();
  __syncthreads();

  // ---- autoregressive decode ----
  for (int t = 0; t < GL; ++t) {
    __syncthreads();
    // embedding: x = dict[token] + pe[t]
    {
      const int b = tid >> 4, j = tid & 15;
      const int tok = s_tok[b];
#pragma unroll
      for (int u = 0; u < 4; ++u) {
        const int d = j * 4 + u;
        s_x[b * DM + d] = P.dict[tok * DM + d] + P.pe[t * DM + d];
      }
    }
    __syncthreads();

    for (int blk = 0; blk < NB; ++blk) {
      const bf16_t* wB = P.wT + (size_t)blk * BLK_STRIDE;

      // ---- self attention ----
      row_ln(s_x, s_h, P.lng[blk * 3 + 0], P.lnb[blk * 3 + 0], tid);
      __syncthreads();
      {
        v16bf af[2];
        load_afrags<DM>(s_h, lane, af);                // hoisted: same A for q,k,v
        for (int tt = wave; tt < 12; tt += NWAVES) {   // q,k,v: 3 x 4 N-tiles
          const int which = tt >> 2, nt = tt & 3;
          v8f d = gemm_tile_pre<DM>(af, wB + which * 4096, nt, lane);
          float* dst = (which == 0) ? s_q : ((which == 1) ? s_k : s_v);
          store_tile(dst, DM, nt, lane, d, (which == 0) ? 0.25f : 1.0f); // q * d_head^-0.5
        }
      }
      __syncthreads();
      // append k,v (position t) to bf16 KV cache
      for (int u = tid; u < BB * DM; u += 256) {
        const int b = u >> 6, c = u & 63;
        const int pair = b * NH + (c >> 4), dd = c & 15;
        const size_t base = (((size_t)blk * 64 + pair) * LS + t) * DH + dd;
        P.kcache[base] = (bf16_t)s_k[u];
        P.vcache[base] = (bf16_t)s_v[u];
      }
      __threadfence();
      __syncthreads();
      // online-softmax attention over positions 0..t (one thread per (b,h))
      if (tid < 64) {
        const int b = tid >> 2, h = tid & 3;
        float qv[DH];
#pragma unroll
        for (int d = 0; d < DH; ++d) qv[d] = s_q[b * DM + h * DH + d];
        const bf16_t* kc = P.kcache + ((size_t)blk * 64 + tid) * LS * DH;
        const bf16_t* vc = P.vcache + ((size_t)blk * 64 + tid) * LS * DH;
        float m = -3.0e38f, l = 0.f, acc[DH];
#pragma unroll
        for (int d = 0; d < DH; ++d) acc[d] = 0.f;
        for (int pos = 0; pos <= t; ++pos) {
          const v16bf kr = *(const v16bf*)(kc + pos * DH);
          const v16bf vr = *(const v16bf*)(vc + pos * DH);
          float s = 0.f;
#pragma unroll
          for (int d = 0; d < DH; ++d) s += qv[d] * (float)kr[d];
          const float mn = fmaxf(m, s);
          const float co = __expf(m - mn), cn = __expf(s - mn);
          l = l * co + cn;
#pragma unroll
          for (int d = 0; d < DH; ++d) acc[d] = acc[d] * co + cn * (float)vr[d];
          m = mn;
        }
        const float inv = 1.0f / l;
#pragma unroll
        for (int d = 0; d < DH; ++d) s_h[b * DM + h * DH + d] = acc[d] * inv;
      }
      __syncthreads();
      {
        v16bf af[2];
        load_afrags<DM>(s_h, lane, af);
        for (int nt = wave; nt < 4; nt += NWAVES) {     // out proj + residual
          v8f d = gemm_tile_pre<DM>(af, wB + OFF_WO, nt, lane);
          store_tile_add(s_x, DM, nt, lane, d);
        }
      }
      __syncthreads();

      // ---- cross attention to slots ----
      row_ln(s_x, s_h, P.lng[blk * 3 + 1], P.lnb[blk * 3 + 1], tid);
      __syncthreads();
      {
        v16bf af[2];
        load_afrags<DM>(s_h, lane, af);
        for (int nt = wave; nt < 4; nt += NWAVES) {
          v8f d = gemm_tile_pre<DM>(af, wB + OFF_CWQ, nt, lane);
          store_tile(s_q, DM, nt, lane, d, 0.25f);
        }
      }
      __syncthreads();
      if (tid < 64) {                                    // 8-slot softmax attn
        const int b = tid >> 2, h = tid & 3;
        const float* kc = P.kcross + ((size_t)blk * 64 + tid) * 8 * DH;
        const float* vc = P.vcross + ((size_t)blk * 64 + tid) * 8 * DH;
        float qv[DH];
#pragma unroll
        for (int d = 0; d < DH; ++d) qv[d] = s_q[b * DM + h * DH + d];
        float sc[8], mx = -3.0e38f;
#pragma unroll
        for (int s = 0; s < 8; ++s) {
          float v = 0.f;
#pragma unroll
          for (int d = 0; d < DH; ++d) v += qv[d] * kc[s * DH + d];
          sc[s] = v;
          mx = fmaxf(mx, v);
        }
        float sum = 0.f;
#pragma unroll
        for (int s = 0; s < 8; ++s) { sc[s] = __expf(sc[s] - mx); sum += sc[s]; }
        const float inv = 1.0f / sum;
#pragma unroll
        for (int d = 0; d < DH; ++d) {
          float o = 0.f;
#pragma unroll
          for (int s = 0; s < 8; ++s) o += sc[s] * vc[s * DH + d];
          s_h[b * DM + h * DH + d] = o * inv;
        }
      }
      __syncthreads();
      {
        v16bf af[2];
        load_afrags<DM>(s_h, lane, af);
        for (int nt = wave; nt < 4; nt += NWAVES) {
          v8f d = gemm_tile_pre<DM>(af, wB + OFF_CWO, nt, lane);
          store_tile_add(s_x, DM, nt, lane, d);
        }
      }
      __syncthreads();

      // ---- FFN ----
      row_ln(s_x, s_h, P.lng[blk * 3 + 2], P.lnb[blk * 3 + 2], tid);
      __syncthreads();
      {
        v16bf af[2];
        load_afrags<DM>(s_h, lane, af);
        for (int nt = wave; nt < 16; nt += NWAVES) {     // fc1: 16x64 @ 64x256
          v8f d = gemm_tile_pre<DM>(af, wB + OFF_FC1, nt, lane);
          store_tile_bias_relu(s_big, FF, nt, lane, d, P.fc1b[blk]);
        }
      }
      __syncthreads();
      {
        v16bf af8[8];
        load_afrags<FF>(s_big, lane, af8);               // 16x256 mid, 8 frags
        for (int nt = wave; nt < 4; nt += NWAVES) {      // fc2: 16x256 @ 256x64
          v8f d = gemm_tile_pre<FF>(af8, wB + OFF_FC2, nt, lane);
          store_tile_add_bias(s_x, DM, nt, lane, d, P.fc2b[blk]);
        }
      }
      __syncthreads();
    }

    // ---- final LN + vocab head ----
    row_ln(s_x, s_h, P.lng[6], P.lnb[6], tid);
    __syncthreads();
    {
      v16bf af[2];
      load_afrags<DM>(s_h, lane, af);                    // built once, reused 16x/wave
      for (int nt = wave; nt < NV / 16; nt += NWAVES) {  // 128 N-tiles
        v8f d = gemm_tile_pre<DM>(af, P.wT + OFF_WOUT, nt, lane);
        const int col = nt * 16 + (lane & 15), rb = (lane >> 4) << 3;
#pragma unroll
        for (int r = 0; r < 8; ++r) P.logits[(rb + r) * NV + col] = d[r];
      }
    }
    __threadfence();
    __syncthreads();

    // ---- per-row max/argmax (first-max tie-break) + log-sum-exp ----
    const int b = tid >> 4, j = tid & 15;
    {
      float mmax = -3.0e38f; int mi = 0;
      for (int v = j; v < NV; v += 16) {
        const float x = P.logits[b * NV + v];
        if (x > mmax) { mmax = x; mi = v; }              // keeps first max in slice
      }
      s_red[tid] = mmax; s_redi[tid] = mi;
      __syncthreads();
      for (int s2 = 8; s2 > 0; s2 >>= 1) {
        if (j < s2) {
          const float o = s_red[tid + s2]; const int oi = s_redi[tid + s2];
          if (o > s_red[tid] || (o == s_red[tid] && oi < s_redi[tid])) {
            s_red[tid] = o; s_redi[tid] = oi;
          }
        }
        __syncthreads();
      }
      if (j == 0) { s_rowm[b] = s_red[tid]; s_rowi[b] = s_redi[tid]; }
      __syncthreads();
      const float mrow = s_rowm[b];
      float sum = 0.f;
      for (int v = j; v < NV; v += 16) sum += __expf(P.logits[b * NV + v] - mrow);
      s_red[tid] = sum;
      __syncthreads();
      for (int s2 = 8; s2 > 0; s2 >>= 1) {
        if (j < s2) s_red[tid] += s_red[tid + s2];
        __syncthreads();
      }
      if (j == 0) s_rowl[b] = __logf(s_red[tid]);
      __syncthreads();
    }
    // outputs + next token
    if (tid < BB) {
      const int am = s_rowi[tid];
      P.out_z[((size_t)tid * NV + am) * GL + t] = 1.0f;  // one-hot (B,VOCAB,16,16)
      s_tok[tid] = am + 1;                               // shift (one-hot col 0 = BOS)
    }
    for (int u = tid; u < BB * NV; u += 256) {
      const int bb = u >> 11, v = u & (NV - 1);
      P.out_logp[((size_t)bb * GL + t) * NV + v] = P.logits[u] - s_rowm[bb] - s_rowl[bb];
    }
    __syncthreads();
  }
}

// ---------------------------------------------------------------------------
// Host launcher
// Input flattening assumption (setup_inputs() dict insertion order, depth
// first): 0 slots, 1 dict_table, 2 pe, 3 w_out, then per block:
//   ln1_g, ln1_b, self.{wq,wk,wv,wo}, ln2_g, ln2_b, cross.{wq,wk,wv,wo},
//   ln3_g, ln3_b, fc1_w, fc1_b, fc2_w, fc2_b     (18 entries x 2 blocks)
// then lnf_g (40), lnf_b (41).
// ---------------------------------------------------------------------------
extern "C" void kernel_launch(void* const* d_in, const int* in_sizes, int n_in,
                              void* d_out, int out_size, void* d_ws, size_t ws_size,
                              hipStream_t stream) {
  (void)in_sizes; (void)out_size; (void)ws_size;
  const float* IN[42];
  for (int i = 0; i < 42; ++i) IN[i] = (i < n_in) ? (const float*)d_in[i] : nullptr;

  // workspace partition (256B aligned)
  char* ws = (char*)d_ws;
  size_t off = 0;
  auto take = [&](size_t bytes) -> char* {
    char* p = ws + off;
    off = (off + bytes + 255) & ~(size_t)255;
    return p;
  };
  bf16_t* wT     = (bf16_t*)take((size_t)(2 * BLK_STRIDE + NV * DM) * sizeof(bf16_t));
  bf16_t* kcache = (bf16_t*)take((size_t)NB * 64 * LS * DH * sizeof(bf16_t));
  bf16_t* vcache = (bf16_t*)take((size_t)NB * 64 * LS * DH * sizeof(bf16_t));
  float*  kcross = (float*) take((size_t)NB * 64 * 8 * DH * sizeof(float));
  float*  vcross = (float*) take((size_t)NB * 64 * 8 * DH * sizeof(float));
  float*  logits = (float*) take((size_t)BB * NV * sizeof(float));

  // 1) convert + transpose all weight matrices into the bf16 arena
  struct TJ { int src; size_t dst; int K, N; };
  TJ jobs[21];
  int nj = 0;
  for (int blk = 0; blk < 2; ++blk) {
    const int base = 4 + blk * 18;
    const size_t wb = (size_t)blk * BLK_STRIDE;
    const int sq[8] = { base + 2, base + 3, base + 4, base + 5,    // self wq,wk,wv,wo
                        base + 8, base + 9, base + 10, base + 11 };// cross wq,wk,wv,wo
    for (int i = 0; i < 8; ++i) jobs[nj++] = { sq[i], wb + (size_t)i * 4096, DM, DM };
    jobs[nj++] = { base + 14, wb + OFF_FC1, DM, FF };  // fc1_w (64,256) -> [256][64]
    jobs[nj++] = { base + 16, wb + OFF_FC2, FF, DM };  // fc2_w (256,64) -> [64][256]
  }
  jobs[nj++] = { 3, (size_t)OFF_WOUT, DM, NV };        // w_out (64,2048) -> [2048][64]
  for (int i = 0; i < nj; ++i) {
    const int n = jobs[i].K * jobs[i].N;
    oc_wt_transpose_bf16<<<(n + 255) / 256, 256, 0, stream>>>(
        IN[jobs[i].src], wT + jobs[i].dst, jobs[i].K, jobs[i].N);
  }

  // 2) zero the one-hot z_gen half of the output
  float* out_z    = (float*)d_out;
  float* out_logp = out_z + (size_t)BB * NV * GL;
  {
    const int n = BB * NV * GL;   // 8,388,608
    oc_zero_f32<<<(n + 255) / 256, 256, 0, stream>>>(out_z, n);
  }

  // 3) persistent single-workgroup decode
  KP kp;
  kp.slots = IN[0]; kp.dict = IN[1]; kp.pe = IN[2];
  for (int blk = 0; blk < 2; ++blk) {
    const int base = 4 + blk * 18;
    kp.lng[blk * 3 + 0] = IN[base + 0];  kp.lnb[blk * 3 + 0] = IN[base + 1];
    kp.lng[blk * 3 + 1] = IN[base + 6];  kp.lnb[blk * 3 + 1] = IN[base + 7];
    kp.lng[blk * 3 + 2] = IN[base + 12]; kp.lnb[blk * 3 + 2] = IN[base + 13];
    kp.fc1b[blk] = IN[base + 15];
    kp.fc2b[blk] = IN[base + 17];
  }
  kp.lng[6] = IN[40]; kp.lnb[6] = IN[41];
  kp.wT = wT;
  kp.kcache = kcache; kp.vcache = vcache;
  kp.kcross = kcross; kp.vcross = vcross;
  kp.logits = logits;
  kp.out_z = out_z; kp.out_logp = out_logp;

  oc_rssm_decode<<<1, 256, 0, stream>>>(kp);
}
